// CausalSelfAttention_66383014527749
// MI455X (gfx1250) — matrix operand order repaired and checked by reading, not compiled, not run on hardware
//
#include <hip/hip_runtime.h>
#include <hip/hip_bf16.h>

// ---------------------------------------------------------------------------
// Types for CDNA5 WMMA (wave32): v_wmma_f32_16x16x32_bf16
// ---------------------------------------------------------------------------
typedef __bf16 bf16_t;
typedef __bf16 v16bf __attribute__((ext_vector_type(16)));
typedef __bf16 v8bf  __attribute__((ext_vector_type(8)));
typedef float  v8f   __attribute__((ext_vector_type(8)));

#define USE_ASYNC_COPY 1

static __device__ __forceinline__ v8f zero8() {
  v8f z = {0.f, 0.f, 0.f, 0.f, 0.f, 0.f, 0.f, 0.f};
  return z;
}

static __device__ __forceinline__ v8f wmma_bf16(v16bf a, v16bf b, v8f c) {
  return __builtin_amdgcn_wmma_f32_16x16x32_bf16(
      false, a, false, b, (short)0, c, false, false);
}

// Copy 16 bytes global -> LDS via CDNA5 async DMA (ASYNCcnt-tracked).
// The low 32 bits of a flat LDS address are the LDS byte offset (ISA aperture
// rules), usable directly as the VDST LDS address.
static __device__ __forceinline__ void cp16(bf16_t* lds, const bf16_t* g) {
#if USE_ASYNC_COPY
  uint32_t l = (uint32_t)(uintptr_t)lds;
  asm volatile("global_load_async_to_lds_b128 %0, %1, off"
               :: "v"(l), "v"(g) : "memory");
#else
  *(v8bf*)lds = *(const v8bf*)g;
#endif
}
static __device__ __forceinline__ void cp_wait() {
#if USE_ASYNC_COPY
  asm volatile("s_wait_asynccnt 0x0" ::: "memory");
#endif
}

// A fragment (16x32 bf16, MxK): lanes 0-15 row M=lane, K={0..7,16..23};
// lanes 16-31 row M=lane-16, K={8..15,24..31}. Tile is row-major [rows][rowLen].
static __device__ __forceinline__ v16bf
load_a_frag(const bf16_t* base, int rowBase, int rowLen, int kBase, int lane) {
  int r = lane & 15, half = lane >> 4;
  const bf16_t* p = base + (rowBase + r) * rowLen + kBase + half * 8;
  union { v16bf v; v8bf h[2]; } u;
  u.h[0] = *(const v8bf*)(p);
  u.h[1] = *(const v8bf*)(p + 16);
  return u.v;
}

// B fragment (32x16 bf16, KxN): lanes 0-15 col N=lane K={0..7,16..23}; lanes 16-31
// col N=lane-16 K={8..15,24..31}. Tile stored [cols][rowLen] with K contiguous.
static __device__ __forceinline__ v16bf
load_b_frag(const bf16_t* base, int colBase, int rowLen, int kBase, int lane) {
  int c = lane & 15, half = lane >> 4;
  const bf16_t* p = base + (colBase + c) * rowLen + kBase + half * 8;
  union { v16bf v; v8bf h[2]; } u;
  u.h[0] = *(const v8bf*)(p);
  u.h[1] = *(const v8bf*)(p + 16);
  return u.v;
}

// ---------------------------------------------------------------------------
// GEMM: block 256(M) x 64(N), K-step 32, 256 threads = 8 waves, each wave
// owns a 32x64 tile (2x4 WMMA tiles -> 8 WMMA per K-step per wave).
// B is PRE-TRANSPOSED (stored [N][K]), so all staging is contiguous 16B
// async-LDS copies. Double-buffered LDS, one barrier per K-step.
// ---------------------------------------------------------------------------
#define BM 256
#define BN 64
#define BK 32

static __device__ __forceinline__ void
stage_a_t(const bf16_t* A, int lda, int m0, int k0, bf16_t* Alds, int tid) {
  const bf16_t* src = A + (size_t)(m0 + tid) * lda + k0;
  bf16_t* dst = Alds + tid * BK;
  cp16(dst, src); cp16(dst + 8, src + 8);
  cp16(dst + 16, src + 16); cp16(dst + 24, src + 24);
}

static __device__ __forceinline__ void
stage_b_t(const bf16_t* BT, int ldb, int n0, int k0, bf16_t* Blds, int tid) {
  int row = tid >> 2, ch = (tid & 3) * 8;
  cp16(Blds + row * BK + ch, BT + (size_t)(n0 + row) * ldb + k0 + ch);
}

static __device__ __forceinline__ void
gemm_core(const bf16_t* A, const bf16_t* BT, int lda, int ldb,
          int m0, int n0, int K, bf16_t* Alds, bf16_t* Blds, v8f (&acc)[2][4]) {
  int tid  = threadIdx.x;
  int lane = tid & 31;
  int wave = tid >> 5;
  int wm = wave * 32;

  stage_a_t(A, lda, m0, 0, Alds, tid);
  stage_b_t(BT, ldb, n0, 0, Blds, tid);
  cp_wait();
  __syncthreads();

  for (int k0 = 0; k0 < K; k0 += BK) {
    int cur = (k0 >> 5) & 1, nxt = cur ^ 1;
    if (k0 + BK < K) {
      stage_a_t(A, lda, m0, k0 + BK, Alds + nxt * (BM * BK), tid);
      stage_b_t(BT, ldb, n0, k0 + BK, Blds + nxt * (BN * BK), tid);
      if (k0 + 2 * BK < K) {
        __builtin_prefetch(A + (size_t)(m0 + tid) * lda + k0 + 2 * BK, 0, 1);
        __builtin_prefetch(BT + (size_t)(n0 + (tid >> 2)) * ldb + k0 + 2 * BK, 0, 1);
      }
    }
    const bf16_t* Ab = Alds + cur * (BM * BK);
    const bf16_t* Bb = Blds + cur * (BN * BK);
    v16bf a0 = load_a_frag(Ab, wm + 0,  BK, 0, lane);
    v16bf a1 = load_a_frag(Ab, wm + 16, BK, 0, lane);
#pragma unroll
    for (int j = 0; j < 4; ++j) {
      v16bf b = load_b_frag(Bb, j * 16, BK, 0, lane);
      acc[0][j] = wmma_bf16(a0, b, acc[0][j]);
      acc[1][j] = wmma_bf16(a1, b, acc[1][j]);
    }
    cp_wait();
    __syncthreads();
  }
}

// ---------------------------------------------------------------------------
// Kernel: fp32 -> bf16 cast (row-major, for x)
// ---------------------------------------------------------------------------
__global__ void cast_f32_bf16(const float* __restrict__ in,
                              bf16_t* __restrict__ out, int n) {
  int i = blockIdx.x * 256 + threadIdx.x;
  int stride = gridDim.x * 256;
  for (; i < n; i += stride) out[i] = (bf16_t)in[i];
}

// ---------------------------------------------------------------------------
// Kernel: tiled transpose + cast: in[z][R][C] fp32 -> out[z][C][R] bf16.
// ---------------------------------------------------------------------------
__global__ __launch_bounds__(256)
void transpose_cast(const float* __restrict__ in, bf16_t* __restrict__ out,
                    int R, int C) {
  __shared__ float t[32][33];
  int c0 = blockIdx.x * 32, r0 = blockIdx.y * 32;
  const float* src = in + (size_t)blockIdx.z * R * C;
  bf16_t* dst = out + (size_t)blockIdx.z * R * C;
  int tx = threadIdx.x, ty = threadIdx.y;  // block (32,8)
#pragma unroll
  for (int i = 0; i < 32; i += 8)
    t[ty + i][tx] = src[(size_t)(r0 + ty + i) * C + (c0 + tx)];
  __syncthreads();
#pragma unroll
  for (int i = 0; i < 32; i += 8)
    dst[(size_t)(c0 + ty + i) * R + (r0 + tx)] = (bf16_t)t[tx][ty + i];
}

// ---------------------------------------------------------------------------
// Kernel: QKV projection. y = x(4096x1024) @ Wkqv[h]^T-stored(2112x1024) + b.
// K/Q/V column boundaries (1024, 2048) are multiples of BN=64, so each block
// writes exactly ONE destination -> the split is uniform in blockIdx.y and
// hoisted out of the store loops. V is stored TRANSPOSED: Vb[bh][dh][n].
// ---------------------------------------------------------------------------
__global__ __launch_bounds__(256)
void qkv_gemm(const bf16_t* __restrict__ xb, const bf16_t* __restrict__ wbT,
              const float* __restrict__ bias,
              bf16_t* __restrict__ Kb, bf16_t* __restrict__ Qb,
              bf16_t* __restrict__ Vb) {
  __shared__ bf16_t Alds[2 * BM * BK];   // 32 KB
  __shared__ bf16_t Blds[2 * BN * BK];   // 8 KB
  int h  = blockIdx.z;
  int m0 = blockIdx.x * BM;
  int n0 = blockIdx.y * BN;
  const bf16_t* BT = wbT + (size_t)h * 1024 * 2112;  // [2112][1024]
  const float* bv = bias + (size_t)h * 2112;

  v8f acc[2][4];
#pragma unroll
  for (int i = 0; i < 2; ++i)
#pragma unroll
    for (int j = 0; j < 4; ++j) acc[i][j] = zero8();

  gemm_core(xb, BT, 1024, 1024, m0, n0, 1024, Alds, Blds, acc);

  int lane = threadIdx.x & 31, wave = threadIdx.x >> 5;
  int half8 = (lane >> 4) * 8, cl = lane & 15;

  if (n0 < 1024) {
    // ---- K block ----
#pragma unroll
    for (int i = 0; i < 2; ++i) {
#pragma unroll
      for (int j = 0; j < 4; ++j) {
        int col = n0 + j * 16 + cl;
        float bcol = bv[col];
#pragma unroll
        for (int r = 0; r < 8; ++r) {
          int m = m0 + wave * 32 + i * 16 + half8 + r;
          size_t bh = (size_t)(m >> 10) * 16 + h;
          Kb[(bh * 1024 + (m & 1023)) * 1024 + col] = (bf16_t)(acc[i][j][r] + bcol);
        }
      }
    }
  } else if (n0 < 2048) {
    // ---- Q block (scale by 1/sqrt(D) = 1/32) ----
#pragma unroll
    for (int i = 0; i < 2; ++i) {
#pragma unroll
      for (int j = 0; j < 4; ++j) {
        int col = n0 + j * 16 + cl;
        float bcol = bv[col];
#pragma unroll
        for (int r = 0; r < 8; ++r) {
          int m = m0 + wave * 32 + i * 16 + half8 + r;
          size_t bh = (size_t)(m >> 10) * 16 + h;
          Qb[(bh * 1024 + (m & 1023)) * 1024 + (col - 1024)] =
              (bf16_t)((acc[i][j][r] + bcol) * 0.03125f);
        }
      }
    }
  } else {
    // ---- V block (single 64-wide block, stored transposed [dh][n]) ----
#pragma unroll
    for (int i = 0; i < 2; ++i) {
#pragma unroll
      for (int j = 0; j < 4; ++j) {
        int dh = j * 16 + cl;                 // n0 == 2048 exactly
        float bcol = bv[2048 + dh];
#pragma unroll
        for (int r = 0; r < 8; ++r) {
          int m = m0 + wave * 32 + i * 16 + half8 + r;
          size_t bh = (size_t)(m >> 10) * 16 + h;
          Vb[(bh * 64 + dh) * 1024 + (m & 1023)] = (bf16_t)(acc[i][j][r] + bcol);
        }
      }
    }
  }
}

// ---------------------------------------------------------------------------
// Kernel: fused causal attention (flash-style). Block = (b, h, 128 rows);
// 8 waves, wave w owns rows [w*16, w*16+16). Double-buffered Q/K staging.
// Per-vgpr-row stats replicated across 16-lane groups (matches C layout).
// ---------------------------------------------------------------------------
__global__ __launch_bounds__(256)
void attn_fused(const bf16_t* __restrict__ Qb, const bf16_t* __restrict__ Kb,
                const bf16_t* __restrict__ VbT, bf16_t* __restrict__ sab) {
  __shared__ bf16_t Qlds[2 * 128 * 32];    // 16 KB
  __shared__ bf16_t Klds[2 * 128 * 32];    // 16 KB
  __shared__ bf16_t Vlds[64 * 128];        // 16 KB, [dh][j]
  __shared__ bf16_t Plds[8 * 16 * 128];    // 32 KB, per-wave [16][128]

  int tid = threadIdx.x, lane = tid & 31, wave = tid >> 5;
  int ib = blockIdx.x;
  int h  = blockIdx.y;
  int b  = blockIdx.z;
  size_t bh = (size_t)b * 16 + h;
  const bf16_t* Qbase = Qb + bh * 1024 * 1024;
  const bf16_t* Kbase = Kb + bh * 1024 * 1024;
  const bf16_t* Vbase = VbT + bh * 64 * 1024;   // [64][1024]

  v8f O[4];
#pragma unroll
  for (int c = 0; c < 4; ++c) O[c] = zero8();
  float mrow[8], lrow[8];
#pragma unroll
  for (int r = 0; r < 8; ++r) { mrow[r] = -1.0e30f; lrow[r] = 0.f; }

  int half8 = (lane >> 4) * 8, cl = lane & 15;
  bf16_t* Pw = Plds + wave * 16 * 128;

  // stager: 128x32 tile, 16 elems (2x16B) per thread
  int srow = tid >> 1, skk = (tid & 1) * 16;

  for (int jb = 0; jb <= ib; ++jb) {
    v8f S[8];
#pragma unroll
    for (int st = 0; st < 8; ++st) S[st] = zero8();

    {
      const bf16_t* qs = Qbase + (size_t)(ib * 128 + srow) * 1024 + skk;
      const bf16_t* ks = Kbase + (size_t)(jb * 128 + srow) * 1024 + skk;
      bf16_t* qd = Qlds + srow * 32 + skk;
      bf16_t* kd = Klds + srow * 32 + skk;
      cp16(qd, qs); cp16(qd + 8, qs + 8);
      cp16(kd, ks); cp16(kd + 8, ks + 8);
    }
    cp_wait();
    __syncthreads();

    for (int dk = 0; dk < 1024; dk += 32) {
      int cur = (dk >> 5) & 1, nxt = cur ^ 1;
      if (dk + 32 < 1024) {
        const bf16_t* qs = Qbase + (size_t)(ib * 128 + srow) * 1024 + dk + 32 + skk;
        const bf16_t* ks = Kbase + (size_t)(jb * 128 + srow) * 1024 + dk + 32 + skk;
        bf16_t* qd = Qlds + nxt * (128 * 32) + srow * 32 + skk;
        bf16_t* kd = Klds + nxt * (128 * 32) + srow * 32 + skk;
        cp16(qd, qs); cp16(qd + 8, qs + 8);
        cp16(kd, ks); cp16(kd + 8, ks + 8);
      }
      const bf16_t* Qc = Qlds + cur * (128 * 32);
      const bf16_t* Kc = Klds + cur * (128 * 32);
      v16bf qa = load_a_frag(Qc, wave * 16, 32, 0, lane);
#pragma unroll
      for (int st = 0; st < 8; ++st) {
        v16bf kf = load_b_frag(Kc, st * 16, 32, 0, lane);
        S[st] = wmma_bf16(qa, kf, S[st]);
      }
      cp_wait();
      __syncthreads();
    }

    // causal mask on diagonal block
    if (jb == ib) {
#pragma unroll
      for (int st = 0; st < 8; ++st) {
        int j_local = st * 16 + cl;
#pragma unroll
        for (int r = 0; r < 8; ++r) {
          int i_local = wave * 16 + half8 + r;
          if (j_local > i_local) S[st][r] = -1.0e30f;
        }
      }
    }

    // online softmax
    float alpha[8];
#pragma unroll
    for (int r = 0; r < 8; ++r) {
      float vmax = S[0][r];
#pragma unroll
      for (int st = 1; st < 8; ++st) vmax = fmaxf(vmax, S[st][r]);
#pragma unroll
      for (int off = 1; off < 16; off <<= 1)
        vmax = fmaxf(vmax, __shfl_xor(vmax, off, 32));
      float mnew = fmaxf(mrow[r], vmax);
      alpha[r] = __expf(mrow[r] - mnew);
      mrow[r] = mnew;
      float rs = 0.f;
#pragma unroll
      for (int st = 0; st < 8; ++st) {
        float p = __expf(S[st][r] - mnew);
        S[st][r] = p;
        rs += p;
      }
#pragma unroll
      for (int off = 1; off < 16; off <<= 1) rs += __shfl_xor(rs, off, 32);
      lrow[r] = lrow[r] * alpha[r] + rs;
    }

#pragma unroll
    for (int c = 0; c < 4; ++c)
#pragma unroll
      for (int r = 0; r < 8; ++r) O[c][r] *= alpha[r];

    // write P (C-layout) to wave-private LDS [16][128]
#pragma unroll
    for (int st = 0; st < 8; ++st)
#pragma unroll
      for (int r = 0; r < 8; ++r)
        Pw[(half8 + r) * 128 + st * 16 + cl] = (bf16_t)S[st][r];

    // stage V_jb from transposed V: contiguous rows, 32 elems per thread
    {
      int dh = tid >> 2, c0 = (tid & 3) * 32;
      const bf16_t* src = Vbase + (size_t)dh * 1024 + jb * 128 + c0;
      bf16_t* dst = Vlds + dh * 128 + c0;
      cp16(dst, src); cp16(dst + 8, src + 8);
      cp16(dst + 16, src + 16); cp16(dst + 24, src + 24);
    }
    cp_wait();
    __syncthreads();

    // O += P @ V
#pragma unroll
    for (int kk = 0; kk < 4; ++kk) {
      v16bf pa = load_a_frag(Pw, 0, 128, kk * 32, lane);
#pragma unroll
      for (int c = 0; c < 4; ++c) {
        v16bf vf = load_b_frag(Vlds, c * 16, 128, kk * 32, lane);
        O[c] = wmma_bf16(pa, vf, O[c]);
      }
    }
    __syncthreads();
  }

  // finalize: O / l -> sa[b, n, h*64 + dh] bf16
#pragma unroll
  for (int c = 0; c < 4; ++c) {
#pragma unroll
    for (int r = 0; r < 8; ++r) {
      int i_local = wave * 16 + half8 + r;
      int n  = ib * 128 + i_local;
      int dh = c * 16 + cl;
      float v = O[c][r] / lrow[r];
      sab[((size_t)b * 1024 + n) * 1024 + h * 64 + dh] = (bf16_t)v;
    }
  }
}

// ---------------------------------------------------------------------------
// Kernel: output projection. out = sa(4096x1024) @ WpT-stored(1024x1024) + bp.
// ---------------------------------------------------------------------------
__global__ __launch_bounds__(256)
void proj_gemm(const bf16_t* __restrict__ sab, const bf16_t* __restrict__ wpT,
               const float* __restrict__ bp, float* __restrict__ out) {
  __shared__ bf16_t Alds[2 * BM * BK];
  __shared__ bf16_t Blds[2 * BN * BK];
  int m0 = blockIdx.x * BM;
  int n0 = blockIdx.y * BN;

  v8f acc[2][4];
#pragma unroll
  for (int i = 0; i < 2; ++i)
#pragma unroll
    for (int j = 0; j < 4; ++j) acc[i][j] = zero8();

  gemm_core(sab, wpT, 1024, 1024, m0, n0, 1024, Alds, Blds, acc);

  int lane = threadIdx.x & 31, wave = threadIdx.x >> 5;
  int half8 = (lane >> 4) * 8, cl = lane & 15;
#pragma unroll
  for (int i = 0; i < 2; ++i) {
#pragma unroll
    for (int j = 0; j < 4; ++j) {
      int col = n0 + j * 16 + cl;
      float bcol = bp[col];
#pragma unroll
      for (int r = 0; r < 8; ++r) {
        int m = m0 + wave * 32 + i * 16 + half8 + r;
        out[(size_t)m * 1024 + col] = acc[i][j][r] + bcol;
      }
    }
  }
}

// ---------------------------------------------------------------------------
// Host launcher
// ---------------------------------------------------------------------------
extern "C" void kernel_launch(void* const* d_in, const int* in_sizes, int n_in,
                              void* d_out, int out_size, void* d_ws, size_t ws_size,
                              hipStream_t stream) {
  (void)in_sizes; (void)n_in; (void)out_size; (void)ws_size;
  const float* x    = (const float*)d_in[0];   // (4,1024,1024)
  const float* Wkqv = (const float*)d_in[1];   // (16,1024,2112)
  const float* bkqv = (const float*)d_in[2];   // (16,2112)
  const float* Wp   = (const float*)d_in[3];   // (1024,1024)
  const float* bp   = (const float*)d_in[4];   // (1024,)
  float* out = (float*)d_out;                  // (4,1024,1024)

  char* ws = (char*)d_ws;
  size_t off = 0;
  bf16_t* xb     = (bf16_t*)(ws + off); off += (size_t)4096 * 1024 * 2;
  bf16_t* wkqvbT = (bf16_t*)(ws + off); off += (size_t)16 * 2112 * 1024 * 2;
  bf16_t* wpbT   = (bf16_t*)(ws + off); off += (size_t)1024 * 1024 * 2;
  bf16_t* Kb     = (bf16_t*)(ws + off); off += (size_t)64 * 1024 * 1024 * 2;
  bf16_t* Qb     = (bf16_t*)(ws + off); off += (size_t)64 * 1024 * 1024 * 2;
  bf16_t* VbT    = (bf16_t*)(ws + off); off += (size_t)64 * 64 * 1024 * 2;
  bf16_t* sab    = (bf16_t*)(ws + off); off += (size_t)4096 * 1024 * 2;

  int n_x = 4096 * 1024;
  cast_f32_bf16<<<(n_x + 255) / 256, 256, 0, stream>>>(x, xb, n_x);
  // Wkqv: [16][1024][2112] -> [16][2112][1024]
  transpose_cast<<<dim3(66, 32, 16), dim3(32, 8), 0, stream>>>(Wkqv, wkqvbT, 1024, 2112);
  // Wp: [1024][1024] -> transposed
  transpose_cast<<<dim3(32, 32, 1), dim3(32, 8), 0, stream>>>(Wp, wpbT, 1024, 1024);

  // QKV projection: M=4096 (16 blocks of 256), N=2112 (33 blocks of 64), 16 heads
  qkv_gemm<<<dim3(16, 33, 16), 256, 0, stream>>>(xb, wkqvbT, bkqv, Kb, Qb, VbT);

  // Fused causal attention
  attn_fused<<<dim3(8, 16, 4), 256, 0, stream>>>(Qb, Kb, VbT, sab);

  // Output projection: M=4096 (16 blocks), N=1024 (16 blocks)
  proj_gemm<<<dim3(16, 16, 1), 256, 0, stream>>>(sab, wpbT, bp, out);
}
